// LogMatryoshkaTXCDR_41686952575010
// MI455X (gfx1250) — matrix operand order repaired and checked by reading, not compiled, not run on hardware
//
#include <hip/hip_runtime.h>
#include <stdint.h>

// Problem constants (from reference)
#define D_IN   512
#define D_SAE  8192
#define T_CTX  32
#define KDIM   (T_CTX * D_IN)   // 16384
#define BATCH  1024
#define KTOP   64

// GEMM tiling
#define BM 128
#define BN 64
#define BK 32
#define MBLK (BATCH / BM)   // 8   (fast grid axis -> W slab reused from L2)
#define NBLK (D_SAE / BN)   // 128 (slow grid axis)

typedef __bf16 v16bf __attribute__((ext_vector_type(16)));
typedef float  v8f   __attribute__((ext_vector_type(8)));
typedef int    v4i   __attribute__((vector_size(16)));   // matches async-LDS builtin params

#if __has_builtin(__builtin_amdgcn_global_load_async_to_lds_b128)
#define HAVE_ASYNC_COPY 1
#else
#define HAVE_ASYNC_COPY 0
#endif

__device__ __forceinline__ uint32_t bf16_rne(float f) {
    uint32_t u = __float_as_uint(f);
    u += 0x7FFFu + ((u >> 16) & 1u);
    return u >> 16;
}
__device__ __forceinline__ float bf16_to_f32(uint32_t h) {
    return __uint_as_float(h << 16);
}
// pack (lo half = even k, hi half = odd k) + residual pack
__device__ __forceinline__ void split2(float a, float b, uint32_t& hi, uint32_t& lo) {
    uint32_t h0 = bf16_rne(a), h1 = bf16_rne(b);
    hi = h0 | (h1 << 16);
    lo = bf16_rne(a - bf16_to_f32(h0)) | (bf16_rne(b - bf16_to_f32(h1)) << 16);
}

struct Frag {
    union { uint4 q[2]; v16bf v; };
};
__device__ __forceinline__ void ld_frag(Frag& f, const uint32_t* lds, int d0, int d1) {
    f.q[0] = *(const uint4*)(lds + d0);
    f.q[1] = *(const uint4*)(lds + d1);
}

__device__ __forceinline__ void wmma3(v8f& c, const Frag& aH, const Frag& aL,
                                      const Frag& bH, const Frag& bL) {
    c = __builtin_amdgcn_wmma_f32_16x16x32_bf16(false, aH.v, false, bH.v, (short)0, c, false, false);
    c = __builtin_amdgcn_wmma_f32_16x16x32_bf16(false, aH.v, false, bL.v, (short)0, c, false, false);
    c = __builtin_amdgcn_wmma_f32_16x16x32_bf16(false, aL.v, false, bH.v, (short)0, c, false, false);
}

// 16B global->LDS copy: async (ASYNCcnt) when toolchain exposes it.
__device__ __forceinline__ void copy_b128(const void* g, void* l) {
#if HAVE_ASYNC_COPY
    __builtin_amdgcn_global_load_async_to_lds_b128((v4i*)g, (v4i*)l, 0, 0);
#else
    *(uint4*)l = *(const uint4*)g;
#endif
}
__device__ __forceinline__ void copy_join() {
#if HAVE_ASYNC_COPY
#if __has_builtin(__builtin_amdgcn_s_wait_asynccnt)
    __builtin_amdgcn_s_wait_asynccnt(0);
#else
    asm volatile("s_wait_asynccnt 0x0" ::: "memory");
#endif
#endif
}

// Shared WMMA core: load fragments from LDS, 12 WMMAs (hi/lo 3-term split).
__device__ __forceinline__ void compute_step(const uint32_t* Ah, const uint32_t* Al,
                                             const uint32_t* Bh, const uint32_t* Bl,
                                             int wm, int wn, int lane, v8f c[2][2]) {
    Frag aH[2], aL[2], bH[2], bL[2];
    const int arow = wm * 32 + (lane & 15);
    const int asel = (lane < 16) ? 0 : 8;   // lane>=16 holds K=16..31
    #pragma unroll
    for (int mt = 0; mt < 2; ++mt) {
        int base = (arow + mt * 16) * 16;
        ld_frag(aH[mt], Ah, base + asel, base + 4 + asel);
        ld_frag(aL[mt], Al, base + asel, base + 4 + asel);
    }
    #pragma unroll
    for (int nt = 0; nt < 2; ++nt) {
        int base = ((wn * 2 + nt) * 32 + lane) * 8;
        ld_frag(bH[nt], Bh, base, base + 4);
        ld_frag(bL[nt], Bl, base, base + 4);
    }
    #pragma unroll
    for (int mt = 0; mt < 2; ++mt)
        #pragma unroll
        for (int nt = 0; nt < 2; ++nt)
            wmma3(c[mt][nt], aH[mt], aL[mt], bH[nt], bL[nt]);
}

__device__ __forceinline__ void store_epilogue(const float* b_enc, float* pre,
                                               int mBase, int nBase, int wm, int wn,
                                               int lane, v8f c[2][2]) {
    #pragma unroll
    for (int nt = 0; nt < 2; ++nt) {
        int col = nBase + wn * 32 + nt * 16 + (lane & 15);
        float bias = b_enc[col];
        #pragma unroll
        for (int mt = 0; mt < 2; ++mt) {
            int rbase = mBase + wm * 32 + mt * 16 + ((lane < 16) ? 0 : 8);
            #pragma unroll
            for (int r = 0; r < 8; ++r)
                pre[(size_t)(rbase + r) * D_SAE + col] = c[mt][nt][r] + bias;
        }
    }
}

// ---------------------------------------------------------------------------
// Pre-split: x -> packed bf16 hi/lo dwords, layout [row][kpair]
// ---------------------------------------------------------------------------
__global__ __launch_bounds__(256) void split_x_kernel(
    const float* __restrict__ x, uint32_t* __restrict__ xh, uint32_t* __restrict__ xl)
{
    const float2* src = (const float2*)x;
    size_t base = (size_t)blockIdx.x * 1024 + threadIdx.x;
    #pragma unroll
    for (int i = 0; i < 4; ++i) {
        size_t p = base + (size_t)i * 256;
        float2 v = src[p];
        split2(v.x, v.y, xh[p], xl[p]);
    }
}

// ---------------------------------------------------------------------------
// Pre-split: W_enc -> fragment-swizzled bf16 tiles
// layout: [kslab(512)][ntile(512)][lane(32)][vgpr(8)] dwords
// ---------------------------------------------------------------------------
__global__ __launch_bounds__(256) void split_w_kernel(
    const float* __restrict__ W, uint32_t* __restrict__ wh, uint32_t* __restrict__ wl)
{
    const int kslab = blockIdx.x;           // 0..511
    const int ng    = blockIdx.y;           // 0..127 (64-col group)
    const int tid   = threadIdx.x;
    const int nl    = tid & 63;
    const int g     = tid >> 6;             // 0..3 -> klocal {0-7,8-15,16-23,24-31}
    const int n     = ng * 64 + nl;
    const int kbase = kslab * 32 + g * 8;

    uint32_t hi[4], lo[4];
    #pragma unroll
    for (int i = 0; i < 4; ++i) {
        float a = W[(size_t)(kbase + 2 * i)     * D_SAE + n];
        float b = W[(size_t)(kbase + 2 * i + 1) * D_SAE + n];
        split2(a, b, hi[i], lo[i]);
    }
    const int ntile = n >> 4;                          // global ntile
    const int lane  = (nl & 15) + ((g >= 2) ? 16 : 0);
    const int vg    = (g & 1) * 4;
    size_t dst = ((size_t)(kslab * 512 + ntile) * 32 + lane) * 8 + vg;
    *(uint4*)(wh + dst) = *(uint4*)hi;
    *(uint4*)(wl + dst) = *(uint4*)lo;
}

// ---------------------------------------------------------------------------
// Encoder GEMM, pre-split operands: staging = pure b128 (async) copies.
// ---------------------------------------------------------------------------
__global__ __launch_bounds__(256) void encoder_wmma_presplit(
    const uint32_t* __restrict__ xh, const uint32_t* __restrict__ xl,
    const uint32_t* __restrict__ wh, const uint32_t* __restrict__ wl,
    const float* __restrict__ b_enc, float* __restrict__ pre)
{
    __shared__ __align__(16) uint32_t Ah[BM * 16], Al[BM * 16];
    __shared__ __align__(16) uint32_t Bh[4 * 32 * 8], Bl[4 * 32 * 8];

    const int tid = threadIdx.x, lane = tid & 31, wave = tid >> 5;
    const int wm = wave >> 1, wn = wave & 1;
    const int mBase = blockIdx.x * BM;   // m fast -> W slab stays in L2
    const int nBase = blockIdx.y * BN;
    const int c0 = tid, c1 = tid + 256;  // A chunk ids (512 x 16B per matrix)

    v8f c[2][2];
    { v8f z = {}; c[0][0] = z; c[0][1] = z; c[1][0] = z; c[1][1] = z; }

    for (int kt = 0; kt < KDIM; kt += BK) {
        const size_t krow = (size_t)(kt >> 1);
        // A tiles: [row][kpair] dwords, 16B chunks
        copy_b128(xh + (size_t)(mBase + (c0 >> 2)) * (KDIM / 2) + krow + (c0 & 3) * 4,
                  &Ah[(c0 >> 2) * 16 + (c0 & 3) * 4]);
        copy_b128(xh + (size_t)(mBase + (c1 >> 2)) * (KDIM / 2) + krow + (c1 & 3) * 4,
                  &Ah[(c1 >> 2) * 16 + (c1 & 3) * 4]);
        copy_b128(xl + (size_t)(mBase + (c0 >> 2)) * (KDIM / 2) + krow + (c0 & 3) * 4,
                  &Al[(c0 >> 2) * 16 + (c0 & 3) * 4]);
        copy_b128(xl + (size_t)(mBase + (c1 >> 2)) * (KDIM / 2) + krow + (c1 & 3) * 4,
                  &Al[(c1 >> 2) * 16 + (c1 & 3) * 4]);
        // B tiles: already fragment-swizzled; 4 local tiles are contiguous
        const size_t tb = ((size_t)(kt >> 5) * 512 + (nBase >> 4)) * 256;
        copy_b128(wh + tb + tid * 4, &Bh[tid * 4]);
        copy_b128(wl + tb + tid * 4, &Bl[tid * 4]);
        copy_join();
        __syncthreads();

        compute_step(Ah, Al, Bh, Bl, wm, wn, lane, c);
        __syncthreads();
    }
    store_epilogue(b_enc, pre, mBase, nBase, wm, wn, lane, c);
}

// ---------------------------------------------------------------------------
// Encoder GEMM, inline-conversion fallback (small workspace).
// ---------------------------------------------------------------------------
__global__ __launch_bounds__(256) void encoder_wmma_inline(
    const float* __restrict__ x, const float* __restrict__ W,
    const float* __restrict__ b_enc, float* __restrict__ pre)
{
    __shared__ __align__(16) uint32_t Ah[BM * 16], Al[BM * 16];
    __shared__ __align__(16) uint32_t Bh[4 * 32 * 8], Bl[4 * 32 * 8];

    const int tid = threadIdx.x, lane = tid & 31, wave = tid >> 5;
    const int wm = wave >> 1, wn = wave & 1;
    const int mBase = blockIdx.x * BM;   // m fast -> W slab stays in L2
    const int nBase = blockIdx.y * BN;

    v8f c[2][2];
    { v8f z = {}; c[0][0] = z; c[0][1] = z; c[1][0] = z; c[1][1] = z; }

    for (int kt = 0; kt < KDIM; kt += BK) {
        // stage A: 128x32 f32 -> (hi, residual) bf16 pairs
        #pragma unroll
        for (int r = 0; r < 8; ++r) {
            int idx = r * 256 + tid;
            int row = idx >> 4, kp = idx & 15;
            const float2 v = *(const float2*)(x + (size_t)(mBase + row) * KDIM + kt + kp * 2);
            split2(v.x, v.y, Ah[row * 16 + kp], Al[row * 16 + kp]);
        }
        // stage B: 32x64 f32 -> fragment-swizzled bf16
        #pragma unroll
        for (int p = 0; p < 4; ++p) {
            int kp = p * 4 + (tid >> 6);
            int n  = tid & 63;
            int kl = kp * 2;
            const float w0 = W[(size_t)(kt + kl)     * D_SAE + nBase + n];
            const float w1 = W[(size_t)(kt + kl + 1) * D_SAE + nBase + n];
            int bl  = (n & 15) + ((kl >= 16) ? 16 : 0);
            int vg  = (kl & 15) >> 1;
            int off = ((n >> 4) * 32 + bl) * 8 + vg;
            split2(w0, w1, Bh[off], Bl[off]);
        }
        if (kt + BK < KDIM) {
            __builtin_prefetch(x + (size_t)(mBase + (tid >> 4)) * KDIM + kt + BK + (tid & 15) * 2, 0, 1);
            __builtin_prefetch(W + (size_t)(kt + BK + (tid >> 6) * 2) * D_SAE + nBase + (tid & 63), 0, 1);
        }
        __syncthreads();

        compute_step(Ah, Al, Bh, Bl, wm, wn, lane, c);
        __syncthreads();
    }
    store_epilogue(b_enc, pre, mBase, nBase, wm, wn, lane, c);
}

// ---------------------------------------------------------------------------
// Top-K: per row, 64 argmax passes out of LDS; writes z and active list.
// ---------------------------------------------------------------------------
__global__ __launch_bounds__(256) void topk_kernel(
    const float* __restrict__ pre, float* __restrict__ z,
    int* __restrict__ actIdx, float* __restrict__ actVal)
{
    __shared__ float buf[D_SAE];
    __shared__ float rv[256];
    __shared__ int   ri[256];
    const int b = blockIdx.x, tid = threadIdx.x;
    const float* row = pre + (size_t)b * D_SAE;
    for (int j = tid; j < D_SAE; j += 256) {
        buf[j] = row[j];
        z[(size_t)b * D_SAE + j] = 0.0f;
    }
    __syncthreads();
    for (int it = 0; it < KTOP; ++it) {
        float best = -__builtin_inff();
        int   bidx = D_SAE;
        for (int j = tid; j < D_SAE; j += 256) {
            float v = buf[j];
            if (v > best || (v == best && j < bidx)) { best = v; bidx = j; }
        }
        rv[tid] = best; ri[tid] = bidx;
        __syncthreads();
        for (int st = 128; st > 0; st >>= 1) {
            if (tid < st) {
                float ov = rv[tid + st]; int oi = ri[tid + st];
                if (ov > rv[tid] || (ov == rv[tid] && oi < ri[tid])) { rv[tid] = ov; ri[tid] = oi; }
            }
            __syncthreads();
        }
        if (tid == 0) {
            int idx = ri[0]; float v = rv[0];
            buf[idx] = -__builtin_inff();
            float rz = v > 0.0f ? v : 0.0f;
            actIdx[b * KTOP + it] = idx;
            actVal[b * KTOP + it] = rz;
            z[(size_t)b * D_SAE + idx] = rz;
        }
        __syncthreads();
    }
}

// ---------------------------------------------------------------------------
// Sparse decoder + loss for one scale. One block per batch row.
// ---------------------------------------------------------------------------
__global__ __launch_bounds__(256) void sparse_decoder_kernel(
    const float* __restrict__ x,
    const float* __restrict__ W_dec, const float* __restrict__ b_dec,
    const int* __restrict__ actIdx, const float* __restrict__ actVal,
    float* __restrict__ loss_out, float* __restrict__ xhat_out,
    int s, int prefix, float lossScale)
{
    __shared__ int   sIdx[KTOP];
    __shared__ float sVal[KTOP];
    __shared__ int   sN;
    __shared__ float red[256];
    const int b = blockIdx.x, tid = threadIdx.x;

    if (tid == 0) {
        int n = 0;
        for (int j = 0; j < KTOP; ++j) {
            int id  = actIdx[b * KTOP + j];
            float v = actVal[b * KTOP + j];
            if (id < prefix && v > 0.0f) { sIdx[n] = id; sVal[n] = v; ++n; }
        }
        sN = n;
    }
    __syncthreads();
    const int n     = sN;
    const int elems = s * D_IN;
    const int start = (T_CTX - s) / 2;

    float partial = 0.0f;
    for (int e = tid; e < elems; e += 256) {
        float acc = b_dec[e];
        for (int j = 0; j < n; ++j)
            acc = fmaf(sVal[j], W_dec[(size_t)sIdx[j] * elems + e], acc);
        int t = e >> 9, d = e & 511;
        float xc  = x[(size_t)b * KDIM + (size_t)(start + t) * D_IN + d];
        float err = acc - xc;
        partial += err * err;
        if (xhat_out) xhat_out[(size_t)b * (T_CTX * D_IN) + e] = acc;
    }
    red[tid] = partial;
    __syncthreads();
    for (int st = 128; st > 0; st >>= 1) {
        if (tid < st) red[tid] += red[tid + st];
        __syncthreads();
    }
    if (tid == 0) atomicAdd(loss_out, red[0] * lossScale);
}

// ---------------------------------------------------------------------------
extern "C" void kernel_launch(void* const* d_in, const int* in_sizes, int n_in,
                              void* d_out, int out_size, void* d_ws, size_t ws_size,
                              hipStream_t stream) {
    (void)in_sizes; (void)n_in; (void)out_size;
    const float* x     = (const float*)d_in[0];
    const float* W_enc = (const float*)d_in[1];
    const float* b_enc = (const float*)d_in[2];
    const float* W_dec[6];
    const float* b_dec[6];
    for (int i = 0; i < 6; ++i) {           // setup_inputs order: W_dec_i, b_dec_i interleaved
        W_dec[i] = (const float*)d_in[3 + 2 * i];
        b_dec[i] = (const float*)d_in[4 + 2 * i];
    }

    float* out      = (float*)d_out;
    float* loss_out = out;                                        // scalar
    float* xhat_out = out + 1;                                    // 1024*32*512
    float* z_out    = out + 1 + (size_t)BATCH * T_CTX * D_IN;     // 1024*8192

    // workspace layout
    const size_t preBytes = (size_t)BATCH * D_SAE * sizeof(float);        // 32 MB
    const size_t actBytes = (size_t)BATCH * KTOP * 8;                     // 0.5 MB
    const size_t xsBytes  = (size_t)BATCH * KDIM * 2;                     // 32 MB (per hi/lo)
    const size_t wsBytes  = (size_t)KDIM * D_SAE * 2;                     // 256 MB (per hi/lo)
    char* wsp = (char*)d_ws;
    float* pre    = (float*)wsp;
    int*   actIdx = (int*)(wsp + preBytes);
    float* actVal = (float*)(wsp + preBytes + actBytes / 2);
    char*  splitBase = wsp + preBytes + actBytes;
    const size_t needSplit = preBytes + actBytes + 2 * xsBytes + 2 * wsBytes;

    (void)hipMemsetAsync(loss_out, 0, sizeof(float), stream);

    dim3 gGemm(MBLK, NBLK);                 // (8, 128): m fast for L2 reuse of W
    if (ws_size >= needSplit) {
        uint32_t* xh = (uint32_t*)splitBase;
        uint32_t* xl = (uint32_t*)(splitBase + xsBytes);
        uint32_t* wh = (uint32_t*)(splitBase + 2 * xsBytes);
        uint32_t* wl = (uint32_t*)(splitBase + 2 * xsBytes + wsBytes);
        split_x_kernel<<<(BATCH * KDIM / 2) / 1024, 256, 0, stream>>>(x, xh, xl);
        split_w_kernel<<<dim3(KDIM / BK, D_SAE / 64), 256, 0, stream>>>(W_enc, wh, wl);
        encoder_wmma_presplit<<<gGemm, 256, 0, stream>>>(xh, xl, wh, wl, b_enc, pre);
    } else {
        encoder_wmma_inline<<<gGemm, 256, 0, stream>>>(x, W_enc, b_enc, pre);
    }

    topk_kernel<<<BATCH, 256, 0, stream>>>(pre, z_out, actIdx, actVal);

    static const int PFX[6] = {1366, 2732, 4097, 5462, 6827, 8192};
    static const int SC[6]  = {1, 2, 4, 8, 16, 32};
    for (int i = 0; i < 6; ++i) {
        float ls = 1.0f / (1024.0f * (float)SC[i] * 6.0f);
        sparse_decoder_kernel<<<BATCH, 256, 0, stream>>>(
            x, W_dec[i], b_dec[i], actIdx, actVal,
            loss_out, (i == 5) ? xhat_out : nullptr, SC[i], PFX[i], ls);
    }
}
